// CrossAttention_1288490189056
// MI455X (gfx1250) — matrix operand order repaired and checked by reading, not compiled, not run on hardware
//
#include <hip/hip_runtime.h>
#include <hip/hip_bf16.h>

typedef _Float16 v16h __attribute__((ext_vector_type(16)));
typedef _Float16 v8h  __attribute__((ext_vector_type(8)));
typedef _Float16 v4h  __attribute__((ext_vector_type(4)));
typedef float    v8f  __attribute__((ext_vector_type(8)));

#define NUM_HEADS 8
#define DHEAD     128
#define INNER     1024
#define TQ        4096
#define TKK       1024
#define BATCH     8
#define DQ        1024
#define DC        768
#define QK_SCALE  0.29730177875068026f  /* 128^-0.25 */
#define FRAG_LD   24   /* padded lane stride (halfs) for swizzled LDS tiles */

// ---------------------------------------------------------------------------
// WMMA helpers (CDNA5 wave32, v_wmma_f32_16x16x32_f16)
// ---------------------------------------------------------------------------
__device__ __forceinline__ v8f wmma16(v16h a, v16h b, v8f c) {
  return __builtin_amdgcn_wmma_f32_16x16x32_f16(
      false, a, false, b, (short)0, c, false, false);
}

__device__ __forceinline__ v8f zero8() {
  v8f z;
#pragma unroll
  for (int i = 0; i < 8; ++i) z[i] = 0.0f;
  return z;
}

__device__ __forceinline__ v16h combine16(v8h lo, v8h hi) {
  v16h r;
#pragma unroll
  for (int t = 0; t < 8; ++t) { r[t] = lo[t]; r[8 + t] = hi[t]; }
  return r;
}

// A fragment (16x32 f16) from row-major global memory.  Per the ISA layout a
// lane's 16 halves are two contiguous 16-byte runs: k in [hg*8, hg*8+8) and
// [16+hg*8, 16+hg*8+8)  ->  2 x global_load_b128.
__device__ __forceinline__ v16h load_a_frag_g(const _Float16* base, int lda) {
  int lane = threadIdx.x & 31;
  int m = lane & 15, hg = lane >> 4;
  const _Float16* p = base + (size_t)m * lda + hg * 8;
  return combine16(*(const v8h*)p, *(const v8h*)(p + 16));
}

// B fragment (32x16) where the source is transposed: element (k,n) =
// base[n*ldb + k].  A lane's 16 halves are two 16-byte runs: k in
// [kh*8, kh*8+8) and [16+kh*8, 16+kh*8+8)  ->  2 x global_load_b128.
__device__ __forceinline__ v16h load_bT_frag_g(const _Float16* base, int ldb) {
  int lane = threadIdx.x & 31;
  int n = lane & 15, kh = lane >> 4;
  const _Float16* p = base + (size_t)n * ldb + kh * 8;
  return combine16(*(const v8h*)p, *(const v8h*)(p + 16));
}

// Fragment from a pre-swizzled LDS tile: 16 contiguous halves per lane
// -> 2 x ds_load_b128, zero repacking.
__device__ __forceinline__ v16h load_frag_lds(const _Float16* p) {
  return combine16(*(const v8h*)p, *(const v8h*)(p + 8));
}

// Swizzle coordinates for the A-fragment layout: element (m, k) of a 16x32
// tile lives at lane = m + 16*((k>>3)&1), slot = (k&7) + 8*((k>>4)&1).
// B-fragment layout: element (k, n) lives at lane = n + 16*((k>>3)&1), slot = (k&7) + 8*((k>>4)&1).

// ---------------------------------------------------------------------------
// Kernel 1: C_f16 = scale * (A_f32 @ B_f32), 64x64 tile / WG, 4 waves.
// TR=true writes C transposed (used for V so P@V B-fragments are contiguous).
// ---------------------------------------------------------------------------
template <bool TR>
__global__ __launch_bounds__(128) void gemm_f32_f16(
    int K, const float* __restrict__ A, int lda,
    const float* __restrict__ B, int ldb,
    _Float16* __restrict__ C, int ldc, float scale) {
  __shared__ _Float16 Asw[4][32][FRAG_LD];  // 4 row-subtiles, fragment order
  __shared__ _Float16 Bsw[4][32][FRAG_LD];  // 4 col-subtiles, fragment order

  int tid = threadIdx.x;
  int wave = tid >> 5, lane = tid & 31;
  int wr = wave >> 1, wc = wave & 1;
  size_t m0 = (size_t)blockIdx.y * 64;
  size_t n0 = (size_t)blockIdx.x * 64;

  v8f acc00 = zero8(), acc01 = zero8(), acc10 = zero8(), acc11 = zero8();

  for (int k0 = 0; k0 < K; k0 += 32) {
    if (k0 + 32 < K) {
      __builtin_prefetch(A + (m0 + (tid >> 1)) * (size_t)lda + k0 + 32, 0, 1);
      __builtin_prefetch(B + (size_t)(k0 + 32 + (tid >> 2)) * ldb + n0 + (tid & 3) * 16, 0, 1);
    }
    // stage A tile 64x32 f32 -> f16, fragment-swizzled (b64 LDS stores)
#pragma unroll
    for (int j = 0; j < 4; ++j) {
      int i = tid + 128 * j;
      int r = i >> 3, c4 = (i & 7) << 2;
      float4 f = *(const float4*)(A + (m0 + r) * (size_t)lda + k0 + c4);
      v4h h;
      h[0] = (_Float16)f.x; h[1] = (_Float16)f.y;
      h[2] = (_Float16)f.z; h[3] = (_Float16)f.w;
      int lp = (r & 15) + 16 * ((c4 >> 3) & 1);
      int s0 = (c4 & 7) + 8 * ((c4 >> 4) & 1);
      *(v4h*)&Asw[r >> 4][lp][s0] = h;
    }
    // stage B tile 32x64 f32 -> f16, fragment-swizzled (b16 scatter)
#pragma unroll
    for (int j = 0; j < 4; ++j) {
      int i = tid + 128 * j;
      int r = i >> 4, c4 = (i & 15) << 2;   // r = k row, c4 = col
      float4 f = *(const float4*)(B + (size_t)(k0 + r) * ldb + n0 + c4);
      int nt = c4 >> 4, nb = (c4 & 15) + 16 * ((r >> 3) & 1), s = (r & 7) + 8 * (r >> 4);
      Bsw[nt][nb + 0][s] = (_Float16)f.x;
      Bsw[nt][nb + 1][s] = (_Float16)f.y;
      Bsw[nt][nb + 2][s] = (_Float16)f.z;
      Bsw[nt][nb + 3][s] = (_Float16)f.w;
    }
    __syncthreads();

    v16h a0 = load_frag_lds(&Asw[wr * 2 + 0][lane][0]);
    v16h a1 = load_frag_lds(&Asw[wr * 2 + 1][lane][0]);
    v16h b0 = load_frag_lds(&Bsw[wc * 2 + 0][lane][0]);
    v16h b1 = load_frag_lds(&Bsw[wc * 2 + 1][lane][0]);
    acc00 = wmma16(a0, b0, acc00);
    acc01 = wmma16(a0, b1, acc01);
    acc10 = wmma16(a1, b0, acc10);
    acc11 = wmma16(a1, b1, acc11);
    __syncthreads();
  }

  int n = lane & 15, mh = (lane >> 4) * 8;
  if (!TR) {
#pragma unroll
    for (int r = 0; r < 8; ++r) {
      size_t row0 = m0 + wr * 32 + mh + r;
      size_t row1 = row0 + 16;
      C[row0 * ldc + n0 + wc * 32 + n]      = (_Float16)(acc00[r] * scale);
      C[row0 * ldc + n0 + wc * 32 + 16 + n] = (_Float16)(acc01[r] * scale);
      C[row1 * ldc + n0 + wc * 32 + n]      = (_Float16)(acc10[r] * scale);
      C[row1 * ldc + n0 + wc * 32 + 16 + n] = (_Float16)(acc11[r] * scale);
    }
  } else {
    // Transposed epilogue: C layout holds 8 consecutive M per lane -> v8h
    // stores.  Rows decompose as (batch = m>>10, key = m&1023); 64-row tiles
    // never cross a batch boundary (1024 % 64 == 0).
    int b = (int)(m0 >> 10);
    size_t key0 = (m0 & 1023) + wr * 32 + mh;
    _Float16* Ct = C + (size_t)b * INNER * TKK;
    size_t col0 = n0 + wc * 32 + n;
    size_t col1 = col0 + 16;
    v8h h00, h01, h10, h11;
#pragma unroll
    for (int r = 0; r < 8; ++r) {
      h00[r] = (_Float16)(acc00[r] * scale);
      h01[r] = (_Float16)(acc01[r] * scale);
      h10[r] = (_Float16)(acc10[r] * scale);
      h11[r] = (_Float16)(acc11[r] * scale);
    }
    *(v8h*)(Ct + col0 * TKK + key0)      = h00;
    *(v8h*)(Ct + col1 * TKK + key0)      = h01;
    *(v8h*)(Ct + col0 * TKK + key0 + 16) = h10;
    *(v8h*)(Ct + col1 * TKK + key0 + 16) = h11;
  }
}

// ---------------------------------------------------------------------------
// Kernel 2: flash attention.  Grid = (Tq/64, heads, batch); 4 waves/WG;
// wave w owns q rows [w*16, w*16+16); online softmax over 64-key chunks.
// K is row-major [key][c] (contiguous c -> bT frags); V is transposed
// [c][key] (contiguous key -> bT frags).
// ---------------------------------------------------------------------------
__global__ __launch_bounds__(128) void attn_kernel(
    const _Float16* __restrict__ Q, const _Float16* __restrict__ Km,
    const _Float16* __restrict__ Vt, _Float16* __restrict__ O) {
  // per-wave P tile stored directly in A-fragment order: [wave][kchunk][lane][slots]
  __shared__ _Float16 Pa[4][2][32][FRAG_LD];

  int tid = threadIdx.x, wave = tid >> 5, lane = tid & 31;
  int n = lane & 15, mh = (lane >> 4) * 8;
  int b = blockIdx.z, h = blockIdx.y;
  size_t q0 = (size_t)blockIdx.x * 64 + wave * 16;

  const _Float16* Qp = Q + ((size_t)b * TQ + q0) * INNER + h * DHEAD;
  const _Float16* Kp = Km + ((size_t)b * TKK) * INNER + h * DHEAD;
  const _Float16* Vp = Vt + (size_t)b * INNER * TKK + (size_t)h * DHEAD * TKK;
  _Float16* Op = O + ((size_t)b * TQ + q0) * INNER + h * DHEAD;

  v16h qa[4];
#pragma unroll
  for (int cc = 0; cc < 4; ++cc) qa[cc] = load_a_frag_g(Qp + cc * 32, INNER);

  v8f oacc[8];
#pragma unroll
  for (int t = 0; t < 8; ++t) oacc[t] = zero8();
  float mrow[8], lrow[8];
#pragma unroll
  for (int r = 0; r < 8; ++r) { mrow[r] = -1e30f; lrow[r] = 0.0f; }

  for (int kc = 0; kc < TKK; kc += 64) {
    if (kc + 64 < TKK)
      __builtin_prefetch(Kp + (size_t)(kc + 64 + (tid >> 1)) * INNER + (tid & 1) * 64, 0, 1);

    // ---- S[16 x 64] = Q @ K^T.  cc-outer: batch 4 independent fragment
    // loads (8 x b128 under one clause), then 4 WMMAs on independent
    // accumulators -> full memory-level parallelism, no per-WMMA waits.
    v8f s[4];
#pragma unroll
    for (int nt = 0; nt < 4; ++nt) s[nt] = zero8();
#pragma unroll
    for (int cc = 0; cc < 4; ++cc) {
      v16h kb[4];
#pragma unroll
      for (int nt = 0; nt < 4; ++nt)
        kb[nt] = load_bT_frag_g(Kp + (size_t)(kc + nt * 16) * INNER + cc * 32, INNER);
#pragma unroll
      for (int nt = 0; nt < 4; ++nt)
        s[nt] = wmma16(qa[cc], kb[nt], s[nt]);
    }

    // ---- online softmax; a C-layout row lives in a 16-lane group
    float alpha[8];
#pragma unroll
    for (int r = 0; r < 8; ++r) {
      float cm = fmaxf(fmaxf(s[0][r], s[1][r]), fmaxf(s[2][r], s[3][r]));
      cm = fmaxf(cm, __shfl_xor(cm, 8, 32));
      cm = fmaxf(cm, __shfl_xor(cm, 4, 32));
      cm = fmaxf(cm, __shfl_xor(cm, 2, 32));
      cm = fmaxf(cm, __shfl_xor(cm, 1, 32));
      float mn = fmaxf(mrow[r], cm);
      alpha[r] = __expf(mrow[r] - mn);
      float rs = 0.0f;
#pragma unroll
      for (int nt = 0; nt < 4; ++nt) {
        float p = __expf(s[nt][r] - mn);
        s[nt][r] = p;
        rs += p;
      }
      rs += __shfl_xor(rs, 8, 32);
      rs += __shfl_xor(rs, 4, 32);
      rs += __shfl_xor(rs, 2, 32);
      rs += __shfl_xor(rs, 1, 32);
      lrow[r] = lrow[r] * alpha[r] + rs;
      mrow[r] = mn;
    }
    // vectorized O rescale (pairs into v_pk_mul_f32)
    v8f av;
#pragma unroll
    for (int r = 0; r < 8; ++r) av[r] = alpha[r];
#pragma unroll
    for (int t = 0; t < 8; ++t) oacc[t] *= av;

    // ---- spill P to LDS *in A-fragment order*: value at (row m, key j) goes
    // to chunk j>>5, lane m + 16*((n>>3)&1), slot (n&7) + 8*(nt&1).
#pragma unroll
    for (int nt = 0; nt < 4; ++nt) {
      int hg = (n >> 3) & 1;
      int slot = (n & 7) + 8 * (nt & 1);
#pragma unroll
      for (int r = 0; r < 8; ++r)
        Pa[wave][nt >> 1][mh + r + 16 * hg][slot] = (_Float16)s[nt][r];
    }
    // ---- O += P @ V: per group, batch 4 fragment loads then 4 independent
    // WMMAs into different oacc tiles.
#pragma unroll
    for (int c2 = 0; c2 < 2; ++c2) {
      v16h pa = load_frag_lds(&Pa[wave][c2][lane][0]);
#pragma unroll
      for (int tg = 0; tg < 2; ++tg) {
        v16h vb[4];
#pragma unroll
        for (int u = 0; u < 4; ++u)
          vb[u] = load_bT_frag_g(Vp + (size_t)((tg * 4 + u) * 16) * TKK + kc + c2 * 32, TKK);
#pragma unroll
        for (int u = 0; u < 4; ++u)
          oacc[tg * 4 + u] = wmma16(pa, vb[u], oacc[tg * 4 + u]);
      }
    }
  }

  // ---- normalize (vectorized 1/l via v_rcp), write f16 O row-major
  v8f iv;
#pragma unroll
  for (int r = 0; r < 8; ++r) iv[r] = __builtin_amdgcn_rcpf(lrow[r]);
#pragma unroll
  for (int t = 0; t < 8; ++t) oacc[t] *= iv;
#pragma unroll
  for (int r = 0; r < 8; ++r)
#pragma unroll
    for (int t = 0; t < 8; ++t)
      Op[(size_t)(mh + r) * INNER + t * 16 + n] = (_Float16)oacc[t][r];
}

// ---------------------------------------------------------------------------
// Kernel 3: out_f32 = A_f16 @ B_f32 + bias.  Same swizzled tiling.
// ---------------------------------------------------------------------------
__global__ __launch_bounds__(128) void gemm_f16_f32_bias(
    int K, const _Float16* __restrict__ A, int lda,
    const float* __restrict__ B, int ldb,
    const float* __restrict__ bias, float* __restrict__ C, int ldc) {
  __shared__ _Float16 Asw[4][32][FRAG_LD];
  __shared__ _Float16 Bsw[4][32][FRAG_LD];

  int tid = threadIdx.x;
  int wave = tid >> 5, lane = tid & 31;
  int wr = wave >> 1, wc = wave & 1;
  size_t m0 = (size_t)blockIdx.y * 64;
  size_t n0 = (size_t)blockIdx.x * 64;

  v8f acc00 = zero8(), acc01 = zero8(), acc10 = zero8(), acc11 = zero8();

  for (int k0 = 0; k0 < K; k0 += 32) {
    if (k0 + 32 < K) {
      __builtin_prefetch(A + (m0 + (tid >> 1)) * (size_t)lda + k0 + 32, 0, 1);
      __builtin_prefetch(B + (size_t)(k0 + 32 + (tid >> 2)) * ldb + n0 + (tid & 3) * 16, 0, 1);
    }
    // stage A (f16): one 16-byte run is exactly 8 fragment slots -> b128 in,
    // b128 out to swizzled LDS.
#pragma unroll
    for (int j = 0; j < 2; ++j) {
      int i = tid + 128 * j;
      int r = i >> 2, c8 = (i & 3) << 3;
      v8h p = *(const v8h*)(A + (m0 + r) * (size_t)lda + k0 + c8);
      int lp = (r & 15) + 16 * ((c8 >> 3) & 1);
      int s0 = 8 * ((c8 >> 4) & 1);
      *(v8h*)&Asw[r >> 4][lp][s0] = p;
    }
#pragma unroll
    for (int j = 0; j < 4; ++j) {
      int i = tid + 128 * j;
      int r = i >> 4, c4 = (i & 15) << 2;
      float4 f = *(const float4*)(B + (size_t)(k0 + r) * ldb + n0 + c4);
      int nt = c4 >> 4, nb = (c4 & 15) + 16 * ((r >> 3) & 1), s = (r & 7) + 8 * (r >> 4);
      Bsw[nt][nb + 0][s] = (_Float16)f.x;
      Bsw[nt][nb + 1][s] = (_Float16)f.y;
      Bsw[nt][nb + 2][s] = (_Float16)f.z;
      Bsw[nt][nb + 3][s] = (_Float16)f.w;
    }
    __syncthreads();

    v16h a0 = load_frag_lds(&Asw[wr * 2 + 0][lane][0]);
    v16h a1 = load_frag_lds(&Asw[wr * 2 + 1][lane][0]);
    v16h b0 = load_frag_lds(&Bsw[wc * 2 + 0][lane][0]);
    v16h b1 = load_frag_lds(&Bsw[wc * 2 + 1][lane][0]);
    acc00 = wmma16(a0, b0, acc00);
    acc01 = wmma16(a0, b1, acc01);
    acc10 = wmma16(a1, b0, acc10);
    acc11 = wmma16(a1, b1, acc11);
    __syncthreads();
  }

  int n = lane & 15, mh = (lane >> 4) * 8;
  float bb0 = bias[n0 + wc * 32 + n];
  float bb1 = bias[n0 + wc * 32 + 16 + n];
#pragma unroll
  for (int r = 0; r < 8; ++r) {
    size_t row0 = m0 + wr * 32 + mh + r;
    size_t row1 = row0 + 16;
    C[row0 * ldc + n0 + wc * 32 + n]      = acc00[r] + bb0;
    C[row0 * ldc + n0 + wc * 32 + 16 + n] = acc01[r] + bb1;
    C[row1 * ldc + n0 + wc * 32 + n]      = acc10[r] + bb0;
    C[row1 * ldc + n0 + wc * 32 + 16 + n] = acc11[r] + bb1;
  }
}

// ---------------------------------------------------------------------------
// Launch: Q/K row-major f16, V transposed f16, flash attention, out proj.
// ---------------------------------------------------------------------------
extern "C" void kernel_launch(void* const* d_in, const int* in_sizes, int n_in,
                              void* d_out, int out_size, void* d_ws, size_t ws_size,
                              hipStream_t stream) {
  (void)in_sizes; (void)n_in; (void)out_size; (void)ws_size;
  const float* x   = (const float*)d_in[0];
  const float* ctx = (const float*)d_in[1];
  const float* Wq  = (const float*)d_in[2];
  const float* Wk  = (const float*)d_in[3];
  const float* Wv  = (const float*)d_in[4];
  const float* Wo  = (const float*)d_in[5];
  const float* bo  = (const float*)d_in[6];
  float* out = (float*)d_out;

  const size_t MQ = (size_t)BATCH * TQ;   // 32768
  const size_t MK = (size_t)BATCH * TKK;  // 8192

  _Float16* qh = (_Float16*)d_ws;         // 64 MB, [B*Tq][inner]
  _Float16* kh = qh + MQ * INNER;         // 16 MB, [B*Tk][inner]
  _Float16* vt = kh + MK * INNER;         // 16 MB, [B][inner][Tk] (transposed)
  _Float16* oh = vt + MK * INNER;         // 64 MB, [B*Tq][inner]

  dim3 blk(128);
  gemm_f32_f16<false><<<dim3(INNER / 64, MQ / 64), blk, 0, stream>>>(
      DQ, x, DQ, Wq, INNER, qh, INNER, QK_SCALE);
  gemm_f32_f16<false><<<dim3(INNER / 64, MK / 64), blk, 0, stream>>>(
      DC, ctx, DC, Wk, INNER, kh, INNER, QK_SCALE);
  gemm_f32_f16<true><<<dim3(INNER / 64, MK / 64), blk, 0, stream>>>(
      DC, ctx, DC, Wv, INNER, vt, TKK, 1.0f);
  attn_kernel<<<dim3(TQ / 64, NUM_HEADS, BATCH), blk, 0, stream>>>(qh, kh, vt, oh);
  gemm_f16_f32_bias<<<dim3(DQ / 64, MQ / 64), blk, 0, stream>>>(
      INNER, oh, INNER, Wo, DQ, bo, out, DQ);
}
